// DualRoPEAttention_12927851561729
// MI455X (gfx1250) — matrix operand order repaired and checked by reading, not compiled, not run on hardware
//
#include <hip/hip_runtime.h>

#define EMBED   1024
#define NHEADS  16
#define HDIM    64
#define SEQ     2048
#define BATCH   2
#define LN_THETA 9.210340371976184f   // ln(10000)

typedef _Float16 v16h  __attribute__((ext_vector_type(16)));
typedef _Float16 half8 __attribute__((ext_vector_type(8)));
typedef float    v8f   __attribute__((ext_vector_type(8)));

union HFrag { v16h v; half8 h[2]; };

// A-matrix fragment (16 x K=32, f16), row-major source, leading dim ldk (halves).
// ISA: lane l -> row l%16; halves 0..7 = K k0..k0+7, halves 8..15 = K k0+16..k0+23, k0 = 8*(l>=16).
__device__ __forceinline__ v16h load_frag_a(const _Float16* p, int ldk) {
  const int l = (int)(threadIdx.x & 31);
  const _Float16* r = p + (l & 15) * ldk + ((l & 16) ? 8 : 0);
  HFrag f;
  f.h[0] = *reinterpret_cast<const half8*>(r);
  f.h[1] = *reinterpret_cast<const half8*>(r + 16);
  return f.v;
}

// B-matrix fragment (K=32 x 16, f16). Each "row" of p (length ldk over K) is one B column.
// ISA: lane l -> col l%16; halves 0..7 = K k0..k0+7, halves 8..15 = K k0+16..k0+23, k0 = 8*(l>=16).
__device__ __forceinline__ v16h load_frag_b(const _Float16* p, int ldk) {
  const int l = (int)(threadIdx.x & 31);
  const _Float16* r = p + (l & 15) * ldk + ((l & 16) ? 8 : 0);
  HFrag f;
  f.h[0] = *reinterpret_cast<const half8*>(r);
  f.h[1] = *reinterpret_cast<const half8*>(r + 16);
  return f.v;
}

__device__ __forceinline__ v8f wmma16(v16h a, v16h b, v8f c) {
  return __builtin_amdgcn_wmma_f32_16x16x32_f16(false, a, false, b, (short)0, c, false, false);
}

// CDNA5 async global->LDS copy (16B per lane), tracked by ASYNCcnt.
// lds_byte_off is the byte offset within the workgroup LDS allocation
// (kernels using this allocate ONLY dynamic shared memory, which starts at 0).
__device__ __forceinline__ void async_b128_to_lds(unsigned lds_byte_off, const void* g) {
  asm volatile("global_load_async_to_lds_b128 %0, %1, off"
               :: "v"(lds_byte_off), "v"(g)
               : "memory");
}
__device__ __forceinline__ void wait_asynccnt0() {
  asm volatile("s_wait_asynccnt 0x0" ::: "memory");
}

// ---------------------------------------------------------------------------
// QKV projection: out = X @ W^T, f32 inputs converted to f16 while staging LDS.
// mode 0/1 (Q/K): fuse RoPE, store f16 as (b, h, s, d).
// mode 2 (V):     store f16 transposed as (b, h, d, s).
// Block: 256 thr (8 waves), tile M=128 (16 rows/wave), N=64, K step 64.
// ---------------------------------------------------------------------------
__global__ __launch_bounds__(256) void qkv_proj_kernel(
    const float* __restrict__ X, const float* __restrict__ W,
    _Float16* __restrict__ out, int mode)
{
  __shared__ _Float16 sA[128 * 64];   // 16 KB
  __shared__ _Float16 sB[64 * 64];    //  8 KB
  const int tid  = (int)threadIdx.x;
  const int wm   = tid >> 5;
  const int lane = tid & 31;
  const int m0   = (int)blockIdx.y * 128;
  const int n0   = (int)blockIdx.x * 64;

  v8f acc[4] = {};

  for (int kb = 0; kb < EMBED; kb += 64) {
#pragma unroll
    for (int i = 0; i < 8; ++i) {              // A tile: 128x64 f32 -> f16
      int g = tid + i * 256;                   // 2048 float4 groups
      int row = g >> 4, c4 = (g & 15) * 4;
      float4 v = *reinterpret_cast<const float4*>(X + (size_t)(m0 + row) * EMBED + kb + c4);
      sA[row * 64 + c4 + 0] = (_Float16)v.x;
      sA[row * 64 + c4 + 1] = (_Float16)v.y;
      sA[row * 64 + c4 + 2] = (_Float16)v.z;
      sA[row * 64 + c4 + 3] = (_Float16)v.w;
    }
#pragma unroll
    for (int i = 0; i < 4; ++i) {              // B tile: 64x64 (rows of W = cols of W^T)
      int g = tid + i * 256;                   // 1024 float4 groups
      int row = g >> 4, c4 = (g & 15) * 4;
      float4 v = *reinterpret_cast<const float4*>(W + (size_t)(n0 + row) * EMBED + kb + c4);
      sB[row * 64 + c4 + 0] = (_Float16)v.x;
      sB[row * 64 + c4 + 1] = (_Float16)v.y;
      sB[row * 64 + c4 + 2] = (_Float16)v.z;
      sB[row * 64 + c4 + 3] = (_Float16)v.w;
    }
    __syncthreads();
#pragma unroll
    for (int kf = 0; kf < 2; ++kf) {
      v16h a = load_frag_a(sA + wm * 1024 + kf * 32, 64);
#pragma unroll
      for (int t = 0; t < 4; ++t) {
        v16h b = load_frag_b(sB + t * 1024 + kf * 32, 64);
        acc[t] = wmma16(a, b, acc[t]);
      }
    }
    __syncthreads();
  }

  const int col = lane & 15;
  const int rh  = (lane >> 4) & 1;
#pragma unroll
  for (int t = 0; t < 4; ++t) {
#pragma unroll
    for (int j = 0; j < 8; ++j) {
      float v = acc[t][j];
      int m = m0 + wm * 16 + j + 8 * rh;       // flat (b,s) row
      int n = n0 + t * 16 + col;               // embed column
      int b = m >> 11, s = m & (SEQ - 1);
      int h = n >> 6,  d = n & 63;
      if (mode < 2) {
        // RoPE: pair partner is adjacent column == adjacent lane in C layout.
        float pv = __shfl_xor(v, 1, 32);
        float ang = (float)s * __expf(-((float)(d & ~1) / (float)HDIM) * LN_THETA);
        float sn, cs;
        __sincosf(ang, &sn, &cs);
        float r = (d & 1) ? (pv * sn + v * cs) : (v * cs - pv * sn);
        out[(((size_t)(b * NHEADS + h)) * SEQ + s) * HDIM + d] = (_Float16)r;
      } else {
        out[(((size_t)(b * NHEADS + h)) * HDIM + d) * SEQ + s] = (_Float16)v;
      }
    }
  }
}

// ---------------------------------------------------------------------------
// Flash attention: grid (S/128, H, B), 8 waves x 16 q-rows.
// K/V tiles staged once per block into LDS with async global->LDS copies
// (shared by all 8 waves), online softmax in f32, P re-layout via per-wave
// LDS buffer. Q,K: (b,h,s,d) f16. VT: (b,h,d,s) f16. O: (b,s,e) f16.
// Dynamic LDS layout (halves): sK [0,2048) | sV [2048,4096) | pbuf [4096,8192).
// ---------------------------------------------------------------------------
__global__ __launch_bounds__(256) void attention_kernel(
    const _Float16* __restrict__ Q, const _Float16* __restrict__ K,
    const _Float16* __restrict__ VT, _Float16* __restrict__ O)
{
  extern __shared__ _Float16 smem[];
  _Float16* sK = smem;                 // 32 k-rows x 64 d   (4 KB)
  _Float16* sV = smem + 2048;          // 64 d-rows x 32 s   (4 KB)
  _Float16* pb = smem + 4096 + (threadIdx.x >> 5) * 512;  // per-wave 16x32 P

  const int tid  = (int)threadIdx.x;
  const int wv   = tid >> 5;
  const int lane = tid & 31;
  const int h = (int)blockIdx.y;
  const int b = (int)blockIdx.z;
  const int q0 = (int)blockIdx.x * 128 + wv * 16;
  const size_t head = (size_t)(b * NHEADS + h);
  const _Float16* qh = Q  + head * SEQ * HDIM;
  const _Float16* kh = K  + head * SEQ * HDIM;
  const _Float16* vh = VT + head * HDIM * SEQ;

  // Q fragments held in registers for the whole loop (d = 0..31, 32..63).
  v16h aq0 = load_frag_a(qh + (size_t)q0 * HDIM + 0,  HDIM);
  v16h aq1 = load_frag_a(qh + (size_t)q0 * HDIM + 32, HDIM);

  v8f acc[4] = {};
  float mrow[8], lrow[8];
#pragma unroll
  for (int j = 0; j < 8; ++j) { mrow[j] = -3.0e38f; lrow[j] = 0.0f; }
  const float scale = 0.125f;   // 1/sqrt(64)

  const int vrow = tid >> 2, vseg = tid & 3;   // V-tile copy coordinates

  for (int k0 = 0; k0 < SEQ; k0 += 32) {
    // Stage K tile (contiguous 4 KB) and V^T tile (64 rows x 64B) via async copies.
    async_b128_to_lds((unsigned)(tid * 16),
                      kh + (size_t)k0 * HDIM + tid * 8);
    async_b128_to_lds((unsigned)(4096 + vrow * 64 + vseg * 16),
                      vh + (size_t)vrow * SEQ + k0 + vseg * 8);
    wait_asynccnt0();
    __syncthreads();

    // scores: 16 q-rows x 32 k-cols, two 16x16 C tiles, K-dim = d (2 chained WMMAs each)
    v8f s0 = {}, s1 = {};
    {
      v16h blo = load_frag_b(sK + 0 * 64 + 0,  64);
      v16h bhi = load_frag_b(sK + 0 * 64 + 32, 64);
      s0 = wmma16(aq0, blo, s0);
      s0 = wmma16(aq1, bhi, s0);
      blo = load_frag_b(sK + 16 * 64 + 0,  64);
      bhi = load_frag_b(sK + 16 * 64 + 32, 64);
      s1 = wmma16(aq0, blo, s1);
      s1 = wmma16(aq1, bhi, s1);
    }
    // online softmax per row (row j+8*(lane>=16); columns live across 16-lane half)
#pragma unroll
    for (int j = 0; j < 8; ++j) {
      float a  = s0[j] * scale;
      float bb = s1[j] * scale;
      float mx = fmaxf(a, bb);
#pragma unroll
      for (int off = 1; off < 16; off <<= 1)
        mx = fmaxf(mx, __shfl_xor(mx, off, 32));
      float mnew  = fmaxf(mrow[j], mx);
      float alpha = __expf(mrow[j] - mnew);
      float p0 = __expf(a  - mnew);
      float p1 = __expf(bb - mnew);
      float rs = p0 + p1;
#pragma unroll
      for (int off = 1; off < 16; off <<= 1)
        rs += __shfl_xor(rs, off, 32);
      lrow[j] = lrow[j] * alpha + rs;
      mrow[j] = mnew;
#pragma unroll
      for (int t = 0; t < 4; ++t) acc[t][j] *= alpha;
      int r = j + 8 * (lane >> 4);
      pb[r * 32 + (lane & 15)]      = (_Float16)p0;
      pb[r * 32 + 16 + (lane & 15)] = (_Float16)p1;
    }
    // P (16x32) as A-fragment, staged V^T columns as B-fragments.
    v16h ap = load_frag_a(pb, 32);
#pragma unroll
    for (int t = 0; t < 4; ++t) {
      v16h bv = load_frag_b(sV + (t * 16) * 32, 32);
      acc[t] = wmma16(ap, bv, acc[t]);
    }
    __syncthreads();   // all waves done with sK/sV before next stage overwrites
  }

  // finalize: divide by row sum, store f16 as (b, s, h*64+d)
#pragma unroll
  for (int j = 0; j < 8; ++j) {
    float inv = 1.0f / lrow[j];
    int s = q0 + j + 8 * (lane >> 4);
#pragma unroll
    for (int t = 0; t < 4; ++t) {
      int d = t * 16 + (lane & 15);
      O[((size_t)b * SEQ + s) * EMBED + h * HDIM + d] = (_Float16)(acc[t][j] * inv);
    }
  }
}

// ---------------------------------------------------------------------------
// Output projection: Y = A @ Wo^T + bo.  A is f16 (4096x1024) from workspace,
// staged to LDS with async copies; W converted f32->f16 in registers.
// Dynamic LDS layout (halves): sA [0,8192) (16 KB) | sB [8192,12288) (8 KB).
// ---------------------------------------------------------------------------
__global__ __launch_bounds__(256) void out_proj_kernel(
    const _Float16* __restrict__ A, const float* __restrict__ W,
    const float* __restrict__ bias, float* __restrict__ Y)
{
  extern __shared__ _Float16 smem[];
  _Float16* sA = smem;
  _Float16* sB = smem + 8192;
  const int tid  = (int)threadIdx.x;
  const int wm   = tid >> 5;
  const int lane = tid & 31;
  const int m0   = (int)blockIdx.y * 128;
  const int n0   = (int)blockIdx.x * 64;

  v8f acc[4] = {};

  for (int kb = 0; kb < EMBED; kb += 64) {
#pragma unroll
    for (int i = 0; i < 4; ++i) {              // A tile: 128x64 f16, async 16B chunks
      int c = tid + i * 256;                   // 1024 chunks
      int row = c >> 3, seg = c & 7;
      async_b128_to_lds((unsigned)(row * 128 + seg * 16),
                        A + (size_t)(m0 + row) * EMBED + kb + seg * 8);
    }
#pragma unroll
    for (int i = 0; i < 4; ++i) {              // B tile: 64x64 f32 -> f16
      int g = tid + i * 256;
      int row = g >> 4, c4 = (g & 15) * 4;
      float4 v = *reinterpret_cast<const float4*>(W + (size_t)(n0 + row) * EMBED + kb + c4);
      sB[row * 64 + c4 + 0] = (_Float16)v.x;
      sB[row * 64 + c4 + 1] = (_Float16)v.y;
      sB[row * 64 + c4 + 2] = (_Float16)v.z;
      sB[row * 64 + c4 + 3] = (_Float16)v.w;
    }
    wait_asynccnt0();
    __syncthreads();
#pragma unroll
    for (int kf = 0; kf < 2; ++kf) {
      v16h a = load_frag_a(sA + wm * 1024 + kf * 32, 64);
#pragma unroll
      for (int t = 0; t < 4; ++t) {
        v16h b = load_frag_b(sB + t * 1024 + kf * 32, 64);
        acc[t] = wmma16(a, b, acc[t]);
      }
    }
    __syncthreads();
  }

  const int col = lane & 15;
  const int rh  = (lane >> 4) & 1;
#pragma unroll
  for (int t = 0; t < 4; ++t) {
#pragma unroll
    for (int j = 0; j < 8; ++j) {
      int m = m0 + wm * 16 + j + 8 * rh;
      int n = n0 + t * 16 + col;
      Y[(size_t)m * EMBED + n] = acc[t][j] + bias[n];
    }
  }
}

// ---------------------------------------------------------------------------
extern "C" void kernel_launch(void* const* d_in, const int* in_sizes, int n_in,
                              void* d_out, int out_size, void* d_ws, size_t ws_size,
                              hipStream_t stream) {
  (void)in_sizes; (void)n_in; (void)out_size; (void)ws_size;
  const float* x  = (const float*)d_in[0];
  const float* Wq = (const float*)d_in[1];
  const float* Wk = (const float*)d_in[2];
  const float* Wv = (const float*)d_in[3];
  const float* Wo = (const float*)d_in[4];
  const float* bo = (const float*)d_in[5];

  const size_t elems = (size_t)BATCH * SEQ * EMBED;   // 4M halves each
  _Float16* qw = (_Float16*)d_ws;
  _Float16* kw = qw + elems;
  _Float16* vw = kw + elems;
  _Float16* aw = vw + elems;                          // attention output, f16 (b,s,e)

  dim3 gproj(EMBED / 64, (BATCH * SEQ) / 128);
  qkv_proj_kernel<<<gproj, 256, 0, stream>>>(x, Wq, qw, 0);
  qkv_proj_kernel<<<gproj, 256, 0, stream>>>(x, Wk, kw, 1);
  qkv_proj_kernel<<<gproj, 256, 0, stream>>>(x, Wv, vw, 2);
  attention_kernel<<<dim3(SEQ / 128, NHEADS, BATCH), 256, 16384, stream>>>(qw, kw, vw, aw);
  out_proj_kernel<<<gproj, 256, 24576, stream>>>(aw, Wo, bo, (float*)d_out);
}
